// SpGAT_41515153883695
// MI455X (gfx1250) — matrix operand order repaired and checked by reading, not compiled, not run on hardware
//
#include <hip/hip_runtime.h>
#include <hip/hip_bf16.h>

// ---------------- problem constants (match reference) ----------------
#define NN    16384      // nodes
#define VDIM  16384      // embedding table rows
#define NE    524288     // edges
#define NH    8          // heads
#define ND    64         // dims per head
#define NEMB  16         // embedding dim
#define NCLS  40
#define FDIM  512        // NH*ND

typedef float  v2f  __attribute__((ext_vector_type(2)));
typedef float  v8f  __attribute__((ext_vector_type(8)));
typedef __bf16 v8bf __attribute__((ext_vector_type(8)));
typedef __bf16 v16bf __attribute__((ext_vector_type(16)));

#define HAVE_TDM __has_builtin(__builtin_amdgcn_tensor_load_to_lds)

#if HAVE_TDM
typedef unsigned int v4u  __attribute__((ext_vector_type(4)));
typedef int          v8i_t __attribute__((ext_vector_type(8)));
typedef int          v4i_t __attribute__((ext_vector_type(4)));

__device__ __forceinline__ unsigned int lds_addr32(const void* p) {
  // LDS aperture: low 32 bits of the flat address are the LDS byte offset
  return (unsigned int)(uintptr_t)p;
}

// Issue a 2D TDM tile load (global -> LDS) with optional per-row LDS padding.
//   esz:  data_size code (0=1B,1=2B,2=4B)
//   td0/td1: tensor dims (elements); t0/t1: tile dims; s0: dim0 stride (elems)
//   pint: pad_interval code (0=2 dw,1=4,...,4=32,...,7=256 dw)
//   pamt: pad_amount code (0=1 dw,...,3=4 dw,...)
__device__ __forceinline__ void tdm_load_2d(unsigned int lds, const void* g,
                                            unsigned int esz, unsigned int td0,
                                            unsigned int td1, unsigned int t0,
                                            unsigned int t1, unsigned int s0,
                                            unsigned int pint,
                                            unsigned int pamt) {
  const unsigned long long ga = (unsigned long long)(uintptr_t)g;
  v4u g0;
  g0[0] = 1u;                                   // count=1 (valid user D#)
  g0[1] = lds;                                  // lds_addr
  g0[2] = (unsigned int)ga;                     // global_addr[31:0]
  g0[3] = (unsigned int)(ga >> 32) | (2u << 30);// global_addr[56:32] | type=2
  v8i_t g1;
  g1[0] = (int)((esz << 16) | (1u << 20) | (pint << 22) | (pamt << 25));
  g1[1] = (int)(td0 << 16);                     // tensor_dim0[15:0] @79:48
  g1[2] = (int)((td0 >> 16) | (td1 << 16));     // tensor_dim0 hi | dim1 lo
  g1[3] = (int)((td1 >> 16) | (t0 << 16));      // tensor_dim1 hi | tile_dim0
  g1[4] = (int)(t1 & 0xffffu);                  // tile_dim1 (tile_dim2=0)
  g1[5] = (int)s0;                              // tensor_dim0_stride[31:0]
  g1[6] = 0;                                    // stride hi | dim1_stride lo
  g1[7] = 0;
  const v4i_t gz4 = {0, 0, 0, 0};               // groups 2/3 unused (2D tile)
  const v8i_t gz8 = {0, 0, 0, 0, 0, 0, 0, 0};
  // 6-arg toolchain form: (g0, g1, g2, g3, extra, cpol)
  __builtin_amdgcn_tensor_load_to_lds(g0, g1, gz4, gz4, gz8, 0);
}
#endif

__device__ __forceinline__ float red16(float v) {
  // reduce across 16-lane half of a wave32 (masks < 16 keep halves separate)
  v += __shfl_xor(v, 1);
  v += __shfl_xor(v, 2);
  v += __shfl_xor(v, 4);
  v += __shfl_xor(v, 8);
  return v;
}

__device__ __forceinline__ float eluf(float x) { return x > 0.f ? x : expm1f(x); }

// ============================================================================
// Stage 1: x_emb_raw += x_in @ emb   (f32 WMMA 16x16x4, K-split over grid.y)
// Double-buffered: TDM streams the next 128x32 x_in tile into LDS (with
// hardware row padding) while the 8 waves run WMMA on the current tile.
// ============================================================================
__global__ __launch_bounds__(256) void k_embed(const float* __restrict__ xin,
                                               const float* __restrict__ emb,
                                               float* __restrict__ xraw) {
  __shared__ float xs[2][128][36];  // row stride 36 floats: 16B align, bank-spread
  __shared__ float es[2][32][16];
  const int tid  = threadIdx.x;
  const int lane = tid & 31;
  const int wv   = tid >> 5;
  const int row0 = blockIdx.x * 128;
  const int kbeg = blockIdx.y * (VDIM / 8);
  const int m  = lane & 15;
  const int kh = (lane >> 4) * 2;
  const int NCH = (VDIM / 8) / 32;   // 64 chunks of K=32

  v8f c;
#pragma unroll
  for (int j = 0; j < 8; ++j) c[j] = 0.0f;

  auto stage_es = [&](int b, int kc) {
    const int k  = tid >> 3;
    const int c2 = (tid & 7) << 1;
    const float2 v = *reinterpret_cast<const float2*>(
        emb + (size_t)(kc + k) * NEMB + c2);
    *reinterpret_cast<float2*>(&es[b][k][c2]) = v;
  };
#if HAVE_TDM
  auto stage_xs = [&](int b, int kc) {
    if (wv == 0)   // one wave drives the Tensor Data Mover; EXEC ignored by TDM
      tdm_load_2d(lds_addr32(&xs[b][0][0]),
                  xin + (size_t)row0 * VDIM + kc,
                  /*esz=4B*/2u, /*td0*/(unsigned)VDIM, /*td1*/(unsigned)NN,
                  /*tile*/32u, 128u, /*stride*/(unsigned)VDIM,
                  /*pad: 4 dw after every 32 dw (one 32-float row)*/4u, 3u);
  };
#else
  auto stage_xs = [&](int b, int kc) {
#pragma unroll
    for (int i = 0; i < 4; ++i) {
      const int idx = tid + 256 * i;
      const int r   = idx >> 3;
      const int c4  = (idx & 7) << 2;
      const float4 v = *reinterpret_cast<const float4*>(
          xin + (size_t)(row0 + r) * VDIM + kc + c4);
      *reinterpret_cast<float4*>(&xs[b][r][c4]) = v;
    }
  };
#endif

  stage_xs(0, kbeg);
  stage_es(0, kbeg);
#if HAVE_TDM
  if (wv == 0) __builtin_amdgcn_s_wait_tensorcnt(0);
#endif
  __syncthreads();

  for (int ic = 0; ic < NCH; ++ic) {
    const int cur = ic & 1;
    const int nxt = cur ^ 1;
    if (ic + 1 < NCH) {                 // prefetch next tile into other buffer
      stage_xs(nxt, kbeg + (ic + 1) * 32);
      stage_es(nxt, kbeg + (ic + 1) * 32);
    }
#pragma unroll
    for (int ks = 0; ks < 8; ++ks) {
      const int kb = ks * 4 + kh;
      v2f a; a.x = xs[cur][wv * 16 + m][kb]; a.y = xs[cur][wv * 16 + m][kb + 1];
      v2f b; b.x = es[cur][kb][m];           b.y = es[cur][kb + 1][m];
      c = __builtin_amdgcn_wmma_f32_16x16x4_f32(false, a, false, b,
                                                (short)0, c, false, false);
    }
#if HAVE_TDM
    if (wv == 0) __builtin_amdgcn_s_wait_tensorcnt(0);
#endif
    __syncthreads();
  }
#pragma unroll
  for (int g = 0; g < 8; ++g) {        // C: VGPR g = rows g / g+8
    const int row = row0 + wv * 16 + g + ((lane >> 4) << 3);
    atomicAdd(&xraw[row * NEMB + m], c[g]);
  }
}

// LayerNorm over 16 dims, one row per 16-lane half-wave, in place
__global__ __launch_bounds__(256) void k_ln16(float* __restrict__ x) {
  const int gw   = (blockIdx.x * blockDim.x + threadIdx.x) >> 5;
  const int lane = threadIdx.x & 31;
  const int row  = gw * 2 + (lane >> 4);
  const int col  = lane & 15;
  const float v  = x[row * NEMB + col];
  const float s  = red16(v);
  const float q  = red16(v * v);
  const float mu  = s * (1.0f / 16.0f);
  const float var = q * (1.0f / 16.0f) - mu * mu;
  x[row * NEMB + col] = (v - mu) * rsqrtf(var + 1e-5f);
}

// ============================================================================
// Per-head GEMM, layer 1 (F_in = 16, f32 WMMA): h = x @ W[h]; s1/s2 fused
// ============================================================================
__global__ __launch_bounds__(256) void k_head16(const float* __restrict__ x,
                                                const float* __restrict__ W,
                                                const float* __restrict__ av,
                                                float* __restrict__ hbuf,
                                                float* __restrict__ s1,
                                                float* __restrict__ s2) {
  __shared__ float xs[16][17];
  const int tid  = threadIdx.x;
  const int lane = tid & 31;
  const int head = tid >> 5;
  const int row0 = blockIdx.x * 16;
  xs[tid >> 4][tid & 15] = x[(size_t)(row0 + (tid >> 4)) * NEMB + (tid & 15)];
  __syncthreads();

  const int m  = lane & 15;
  const int kh = (lane >> 4) * 2;
  const float* Wh = W + (size_t)head * NEMB * ND;
  v8f c[4];
#pragma unroll
  for (int i = 0; i < 4; ++i)
#pragma unroll
    for (int j = 0; j < 8; ++j) c[i][j] = 0.0f;

#pragma unroll
  for (int ks = 0; ks < 4; ++ks) {
    const int kb = ks * 4 + kh;
    v2f a; a.x = xs[m][kb]; a.y = xs[m][kb + 1];
#pragma unroll
    for (int nt = 0; nt < 4; ++nt) {
      v2f b; b.x = Wh[kb * ND + nt * 16 + m];
             b.y = Wh[(kb + 1) * ND + nt * 16 + m];
      c[nt] = __builtin_amdgcn_wmma_f32_16x16x4_f32(false, a, false, b,
                                                    (short)0, c[nt], false, false);
    }
  }
  const float* ah = av + head * 2 * ND;
#pragma unroll
  for (int g = 0; g < 8; ++g) {
    const int row = row0 + g + ((lane >> 4) << 3);
    float q1 = 0.f, q2 = 0.f;
#pragma unroll
    for (int nt = 0; nt < 4; ++nt) {
      const float v = c[nt][g];
      hbuf[((size_t)head * NN + row) * ND + nt * 16 + m] = v;
      q1 += v * ah[nt * 16 + m];
      q2 += v * ah[ND + nt * 16 + m];
    }
    q1 = red16(q1);
    q2 = red16(q2);
    if (m == 0) { s1[head * NN + row] = q1; s2[head * NN + row] = q2; }
  }
}

// W (H,512,64) f32 -> Wbt (H,64,512) bf16 (transposed for contiguous B frags)
__global__ __launch_bounds__(256) void k_cvt_w(const float* __restrict__ W,
                                               __bf16* __restrict__ Wbt) {
  const int idx  = blockIdx.x * 256 + threadIdx.x;   // NH*ND*FDIM
  const int head = idx >> 15;                        // ND*FDIM = 32768
  const int rem  = idx & 32767;
  const int n    = rem >> 9;
  const int k    = rem & 511;
  Wbt[idx] = (__bf16)W[((size_t)head << 15) + k * ND + n];
}

// ============================================================================
// Per-head GEMM, layers 2/f (F_in = 512, bf16 WMMA 16x16x32, f32 accumulate)
// Activation tile (16x512 bf16) staged via TDM with hardware row padding.
// ============================================================================
__global__ __launch_bounds__(256) void k_head512(const __bf16* __restrict__ xb,
                                                 const __bf16* __restrict__ Wbt,
                                                 const float* __restrict__ av,
                                                 float* __restrict__ hbuf,
                                                 float* __restrict__ s1,
                                                 float* __restrict__ s2) {
  __shared__ __bf16 xs[16][FDIM + 8];   // +8 bf16 (= 4 dw) per row
  const int tid  = threadIdx.x;
  const int lane = tid & 31;
  const int head = tid >> 5;
  const int row0 = blockIdx.x * 16;
#if HAVE_TDM
  if (tid < 32) {
    tdm_load_2d(lds_addr32(&xs[0][0]), xb + (size_t)row0 * FDIM,
                /*esz=2B*/1u, /*td0*/(unsigned)FDIM, /*td1*/(unsigned)NN,
                /*tile*/(unsigned)FDIM, 16u, /*stride*/(unsigned)FDIM,
                /*pad: 4 dw after every 256 dw (one 512-bf16 row)*/7u, 3u);
    __builtin_amdgcn_s_wait_tensorcnt(0);
  }
#else
#pragma unroll
  for (int i = 0; i < 4; ++i) {          // stage 16x512 bf16 row tile
    const int idx = tid + 256 * i;
    const int r   = idx >> 6;
    const int c8  = (idx & 63) << 3;
    *reinterpret_cast<v8bf*>(&xs[r][c8]) =
        *reinterpret_cast<const v8bf*>(xb + (size_t)(row0 + r) * FDIM + c8);
  }
#endif
  __syncthreads();

  const int m     = lane & 15;
  const int koffA = (lane >> 4) << 3;   // A: K runs {koffA..+8, 16+koffA..+8}
  const int koffB = (lane >> 4) << 3;   // B: K runs {koffB..+8, 16+koffB..+8}
  v8f c[4];
#pragma unroll
  for (int i = 0; i < 4; ++i)
#pragma unroll
    for (int j = 0; j < 8; ++j) c[i][j] = 0.0f;

  const __bf16* Bbase = Wbt + (size_t)head * ND * FDIM;
#pragma unroll
  for (int ks = 0; ks < 16; ++ks) {
    const int kb = ks * 32;
    const v8bf alo = *reinterpret_cast<const v8bf*>(&xs[m][kb + koffA]);
    const v8bf ahi = *reinterpret_cast<const v8bf*>(&xs[m][kb + 16 + koffA]);
    const v16bf a = __builtin_shufflevector(alo, ahi, 0, 1, 2, 3, 4, 5, 6, 7,
                                            8, 9, 10, 11, 12, 13, 14, 15);
#pragma unroll
    for (int nt = 0; nt < 4; ++nt) {
      const __bf16* bp = Bbase + (size_t)(nt * 16 + m) * FDIM + kb + koffB;
      const v8bf blo = *reinterpret_cast<const v8bf*>(bp);
      const v8bf bhi = *reinterpret_cast<const v8bf*>(bp + 16);
      const v16bf b = __builtin_shufflevector(blo, bhi, 0, 1, 2, 3, 4, 5, 6, 7,
                                              8, 9, 10, 11, 12, 13, 14, 15);
      c[nt] = __builtin_amdgcn_wmma_f32_16x16x32_bf16(false, a, false, b,
                                                      (short)0, c[nt],
                                                      false, false);
    }
  }
  const float* ah = av + head * 2 * ND;
#pragma unroll
  for (int g = 0; g < 8; ++g) {
    const int row = row0 + g + ((lane >> 4) << 3);
    float q1 = 0.f, q2 = 0.f;
#pragma unroll
    for (int nt = 0; nt < 4; ++nt) {
      const float v = c[nt][g];
      hbuf[((size_t)head * NN + row) * ND + nt * 16 + m] = v;
      q1 += v * ah[nt * 16 + m];
      q2 += v * ah[ND + nt * 16 + m];
    }
    q1 = red16(q1);
    q2 = red16(q2);
    if (m == 0) { s1[head * NN + row] = q1; s2[head * NN + row] = q2; }
  }
}

// ============================================================================
// Edge aggregation: exploits sorted edge_row — register-segmented reduction,
// flush with atomics only on row change. One wave per (head, 256-edge segment),
// each lane owns 2 feature dims.
// ============================================================================
__global__ __launch_bounds__(256) void k_edge(const int* __restrict__ erow,
                                              const int* __restrict__ ecol,
                                              const float* __restrict__ s1,
                                              const float* __restrict__ s2,
                                              const float* __restrict__ hbuf,
                                              float* __restrict__ acc,
                                              float* __restrict__ rowsum) {
  const int head = blockIdx.y;
  const int lane = threadIdx.x & 31;
  const int seg  = blockIdx.x * 8 + (threadIdx.x >> 5);
  const int e0   = seg * 256;
  if (e0 >= NE) return;
  const int e1 = (e0 + 256 > NE) ? NE : (e0 + 256);

  const float* hf  = hbuf + (size_t)head * NN * ND;
  const float* s1h = s1 + head * NN;
  const float* s2h = s2 + head * NN;

  int   cur = erow[e0];
  float a0 = 0.f, a1v = 0.f, ws = 0.f;
  for (int e = e0; e < e1; ++e) {
    const int r  = erow[e];
    const int cn = ecol[e];
    if (r != cur) {                        // uniform branch across the wave
      atomicAdd(&acc[(size_t)cur * FDIM + head * ND + lane * 2 + 0], a0);
      atomicAdd(&acc[(size_t)cur * FDIM + head * ND + lane * 2 + 1], a1v);
      if (lane == 0) atomicAdd(&rowsum[head * NN + cur], ws);
      a0 = a1v = ws = 0.f;
      cur = r;
    }
    if (e + 8 < e1)                        // global_prefetch_b8 on gfx1250
      __builtin_prefetch(hf + (size_t)ecol[e + 8] * ND + lane * 2, 0, 1);
    const float z  = s1h[r] + s2h[cn];
    const float lr = z > 0.f ? z : 0.2f * z;
    const float w  = __expf(-lr);
    const float2 hv =
        *reinterpret_cast<const float2*>(hf + (size_t)cn * ND + lane * 2);
    a0  += w * hv.x;
    a1v += w * hv.y;
    ws  += w;
  }
  atomicAdd(&acc[(size_t)cur * FDIM + head * ND + lane * 2 + 0], a0);
  atomicAdd(&acc[(size_t)cur * FDIM + head * ND + lane * 2 + 1], a1v);
  if (lane == 0) atomicAdd(&rowsum[head * NN + cur], ws);
}

// ============================================================================
// Finalize: out = LN( elu?( acc/rowsum ) ), optional ELU after LN (final layer),
// optional bf16 downconvert for next layer. One block per node, in place.
// ============================================================================
__global__ __launch_bounds__(256) void k_finalize(const float* __restrict__ rowsum,
                                                  float* __restrict__ xout,
                                                  __bf16* __restrict__ xbout,
                                                  int elu_in, int elu_out) {
  const int i    = blockIdx.x;
  const int t    = threadIdx.x;
  const int lane = t & 31;
  const int wid  = t >> 5;
  __shared__ float rs[NH];
  __shared__ float shs[8], shq[8];
  if (t < NH) rs[t] = rowsum[t * NN + i];
  __syncthreads();

  float v[2];
#pragma unroll
  for (int j = 0; j < 2; ++j) {
    const int col = t + j * 256;
    float u = xout[(size_t)i * FDIM + col] / rs[col >> 6];
    if (elu_in) u = eluf(u);
    v[j] = u;
  }
  float s = v[0] + v[1];
  float q = v[0] * v[0] + v[1] * v[1];
#pragma unroll
  for (int off = 16; off >= 1; off >>= 1) {
    s += __shfl_xor(s, off);
    q += __shfl_xor(q, off);
  }
  if (lane == 0) { shs[wid] = s; shq[wid] = q; }
  __syncthreads();
  if (t == 0) {
    float S = 0.f, Q = 0.f;
    for (int w = 0; w < 8; ++w) { S += shs[w]; Q += shq[w]; }
    const float mu  = S * (1.0f / FDIM);
    const float var = Q * (1.0f / FDIM) - mu * mu;
    shs[0] = mu;
    shq[0] = rsqrtf(var + 1e-5f);
  }
  __syncthreads();
  const float mu = shs[0], r = shq[0];
#pragma unroll
  for (int j = 0; j < 2; ++j) {
    const int col = t + j * 256;
    float o = (v[j] - mu) * r;
    if (elu_out) o = eluf(o);
    xout[(size_t)i * FDIM + col] = o;
    if (xbout) xbout[(size_t)i * FDIM + col] = (__bf16)o;
  }
}

// ============================================================================
// Classifier: logits = x @ Wo + bo, then log_softmax over 40 classes.
// ============================================================================
__global__ __launch_bounds__(128) void k_classifier(const float* __restrict__ x,
                                                    const float* __restrict__ Wo,
                                                    const float* __restrict__ bo,
                                                    float* __restrict__ out) {
  const int i = blockIdx.x;
  const int t = threadIdx.x;
  __shared__ float xr[FDIM];
  __shared__ float logits[NCLS];
  __shared__ float mred, sred;
  for (int c = t; c < FDIM; c += 128) xr[c] = x[(size_t)i * FDIM + c];
  __syncthreads();
  if (t < NCLS) {
    float a = bo[t];
    for (int k = 0; k < FDIM; ++k) a += xr[k] * Wo[k * NCLS + t];
    logits[t] = a;
  }
  __syncthreads();
  if (t == 0) {
    float mx = -1e30f;
    for (int c = 0; c < NCLS; ++c) mx = fmaxf(mx, logits[c]);
    float s = 0.f;
    for (int c = 0; c < NCLS; ++c) s += __expf(logits[c] - mx);
    mred = mx;
    sred = logf(s);
  }
  __syncthreads();
  if (t < NCLS) out[(size_t)i * NCLS + t] = logits[t] - mred - sred;
}

// ============================================================================
extern "C" void kernel_launch(void* const* d_in, const int* in_sizes, int n_in,
                              void* d_out, int out_size, void* d_ws,
                              size_t ws_size, hipStream_t stream) {
  (void)in_sizes; (void)n_in; (void)out_size; (void)ws_size;
  const float* x_in = (const float*)d_in[0];
  const float* emb  = (const float*)d_in[1];
  const float* W1   = (const float*)d_in[2];
  const float* a1   = (const float*)d_in[3];
  const float* W2   = (const float*)d_in[4];
  const float* a2   = (const float*)d_in[5];
  const float* Wf   = (const float*)d_in[6];
  const float* af   = (const float*)d_in[7];
  const float* Wo   = (const float*)d_in[8];
  const float* bo   = (const float*)d_in[9];
  const int*   erow = (const int*)d_in[10];
  const int*   ecol = (const int*)d_in[11];
  float* out = (float*)d_out;

  char* ws = (char*)d_ws;
  size_t off = 0;
  auto take = [&](size_t bytes) -> char* {
    char* p = ws + off;
    off = (off + bytes + 255) & ~(size_t)255;
    return p;
  };
  float*  xemb = (float*)take((size_t)NN * NEMB * 4);
  float*  buf0 = (float*)take((size_t)NN * FDIM * 4);
  float*  buf1 = (float*)take((size_t)NN * FDIM * 4);
  float*  hbuf = (float*)take((size_t)NN * FDIM * 4);
  float*  s1   = (float*)take((size_t)NH * NN * 4);
  float*  s2   = (float*)take((size_t)NH * NN * 4);
  float*  rsum = (float*)take((size_t)NH * NN * 4);
  __bf16* xb   = (__bf16*)take((size_t)NN * FDIM * 2);
  __bf16* Wbt  = (__bf16*)take((size_t)NH * FDIM * ND * 2);

  // ---- stage 1: embedding GEMM (f32 WMMA + TDM double buffering) + LN ----
  (void)hipMemsetAsync(xemb, 0, (size_t)NN * NEMB * 4, stream);
  k_embed<<<dim3(NN / 128, 8), 256, 0, stream>>>(x_in, emb, xemb);
  k_ln16<<<NN / 16, 256, 0, stream>>>(xemb);

  // ---- GAT layer 1 (F_in = 16, f32 WMMA) ----
  (void)hipMemsetAsync(buf0, 0, (size_t)NN * FDIM * 4, stream);
  (void)hipMemsetAsync(rsum, 0, (size_t)NH * NN * 4, stream);
  k_head16<<<NN / 16, 256, 0, stream>>>(xemb, W1, a1, hbuf, s1, s2);
  k_edge<<<dim3(NE / 2048, NH), 256, 0, stream>>>(erow, ecol, s1, s2, hbuf,
                                                  buf0, rsum);
  k_finalize<<<NN, 256, 0, stream>>>(rsum, buf0, xb, 1, 0);

  // ---- GAT layer 2 (F_in = 512, bf16 WMMA, TDM-staged activations) ----
  k_cvt_w<<<(NH * FDIM * ND) / 256, 256, 0, stream>>>(W2, Wbt);
  (void)hipMemsetAsync(buf1, 0, (size_t)NN * FDIM * 4, stream);
  (void)hipMemsetAsync(rsum, 0, (size_t)NH * NN * 4, stream);
  k_head512<<<NN / 16, 256, 0, stream>>>(xb, Wbt, a2, hbuf, s1, s2);
  k_edge<<<dim3(NE / 2048, NH), 256, 0, stream>>>(erow, ecol, s1, s2, hbuf,
                                                  buf1, rsum);
  k_finalize<<<NN, 256, 0, stream>>>(rsum, buf1, xb, 1, 0);

  // ---- final GAT layer (no ELU inside; LN then ELU) ----
  k_cvt_w<<<(NH * FDIM * ND) / 256, 256, 0, stream>>>(Wf, Wbt);
  (void)hipMemsetAsync(buf0, 0, (size_t)NN * FDIM * 4, stream);
  (void)hipMemsetAsync(rsum, 0, (size_t)NH * NN * 4, stream);
  k_head512<<<NN / 16, 256, 0, stream>>>(xb, Wbt, af, hbuf, s1, s2);
  k_edge<<<dim3(NE / 2048, NH), 256, 0, stream>>>(erow, ecol, s1, s2, hbuf,
                                                  buf0, rsum);
  k_finalize<<<NN, 256, 0, stream>>>(rsum, buf0, (__bf16*)nullptr, 0, 1);

  // ---- classifier + log-softmax ----
  k_classifier<<<NN, 128, 0, stream>>>(buf0, Wo, bo, out);
}
